// OrderCourierHeteroGNN_71433896067324
// MI455X (gfx1250) — compile-verified, matrix operand order checked by reading, not run on hardware
//
#include <hip/hip_runtime.h>
#include <hip/hip_bf16.h>

#define DIM 128

typedef __attribute__((ext_vector_type(16))) __bf16 v16bf;
typedef __attribute__((ext_vector_type(8)))  __bf16 v8bf;
typedef __attribute__((ext_vector_type(8)))  float  v8f;

// ---------- helpers ----------
__device__ __forceinline__ unsigned f2ord(float f) {
    unsigned u = __float_as_uint(f);
    return (u & 0x80000000u) ? ~u : (u | 0x80000000u);
}
__device__ __forceinline__ float ord2f(unsigned u) {
    return (u & 0x80000000u) ? __uint_as_float(u & 0x7fffffffu)
                             : __uint_as_float(~u);
}

// ---------- elementwise ----------
__global__ void zero_kernel(float* __restrict__ p, long n) {
    long i = (long)blockIdx.x * blockDim.x + threadIdx.x;
    if (i < n) p[i] = 0.0f;
}

__global__ void relu_kernel(const float* __restrict__ x, float* __restrict__ y, long n) {
    long i = (long)blockIdx.x * blockDim.x + threadIdx.x;
    if (i < n) y[i] = fmaxf(x[i], 0.0f);
}

// ---------- H[M x 128] = X[M x 128] @ W[128 x 128], bf16x3 split for ~fp32 accuracy ----------
// Block: 256 threads = 8 waves; each wave owns a 16-row strip, sweeping 8 N-tiles.
// W is staged in LDS transposed [n][k] as bf16 hi/lo so B-fragments are 16B ds loads.
__global__ void __launch_bounds__(256)
gemm_bf16x3_kernel(const float* __restrict__ X, const float* __restrict__ W,
                   float* __restrict__ H, int M) {
    __shared__ __bf16 sBhi[DIM][DIM];   // [n][k], 32 KB
    __shared__ __bf16 sBlo[DIM][DIM];   // [n][k], 32 KB

    for (int i = threadIdx.x; i < DIM * DIM; i += 256) {
        int k = i >> 7, n = i & 127;
        float w = W[i];
        __bf16 whi = (__bf16)w;
        sBhi[n][k] = whi;
        sBlo[n][k] = (__bf16)(w - (float)whi);
    }
    __syncthreads();

    const int wave = threadIdx.x >> 5;
    const int lane = threadIdx.x & 31;
    const int hh   = lane >> 4;        // half of wave (selects K sub-range per ISA layout)
    const int ln   = lane & 15;
    const int row0 = blockIdx.x * 128 + wave * 16;
    const int m    = row0 + ln;        // A-fragment row for this lane
    const bool valid = (m < M);

    // A fragments: 4 K-chunks of 32, hi/lo bf16 split.
    v16bf ahi[4], alo[4];
    const float4 z4 = make_float4(0.f, 0.f, 0.f, 0.f);
    const float4* X4 = (const float4*)X;
#pragma unroll
    for (int kc = 0; kc < 4; ++kc) {
        const int kbase = kc * 32 + 8 * hh;                 // elements 0..7  -> K=kbase..kbase+7
        float4 x0 = valid ? X4[(size_t)m * 32 + (kbase >> 2)]            : z4;
        float4 x1 = valid ? X4[(size_t)m * 32 + (kbase >> 2) + 1]        : z4;
        float4 x2 = valid ? X4[(size_t)m * 32 + ((kbase + 16) >> 2)]     : z4;  // elements 8..15
        float4 x3 = valid ? X4[(size_t)m * 32 + ((kbase + 16) >> 2) + 1] : z4;
        const float xs[16] = { x0.x, x0.y, x0.z, x0.w,  x1.x, x1.y, x1.z, x1.w,
                               x2.x, x2.y, x2.z, x2.w,  x3.x, x3.y, x3.z, x3.w };
#pragma unroll
        for (int e = 0; e < 16; ++e) {
            __bf16 h = (__bf16)xs[e];
            ahi[kc][e] = h;
            alo[kc][e] = (__bf16)(xs[e] - (float)h);
        }
    }

#pragma unroll
    for (int nt = 0; nt < 8; ++nt) {
        const int n = nt * 16 + ln;     // B-fragment column for this lane
        v8f acc = {};
#pragma unroll
        for (int kc = 0; kc < 4; ++kc) {
            const int kbase = kc * 32 + 8 * hh;
            v8bf bh0 = *(const v8bf*)&sBhi[n][kbase];
            v8bf bh1 = *(const v8bf*)&sBhi[n][kbase + 16];
            v8bf bl0 = *(const v8bf*)&sBlo[n][kbase];
            v8bf bl1 = *(const v8bf*)&sBlo[n][kbase + 16];
            v16bf bhi = __builtin_shufflevector(bh0, bh1, 0,1,2,3,4,5,6,7,8,9,10,11,12,13,14,15);
            v16bf blo = __builtin_shufflevector(bl0, bl1, 0,1,2,3,4,5,6,7,8,9,10,11,12,13,14,15);
            // small terms first, then the main product
            acc = __builtin_amdgcn_wmma_f32_16x16x32_bf16(false, alo[kc], false, bhi, (short)0, acc, false, false);
            acc = __builtin_amdgcn_wmma_f32_16x16x32_bf16(false, ahi[kc], false, blo, (short)0, acc, false, false);
            acc = __builtin_amdgcn_wmma_f32_16x16x32_bf16(false, ahi[kc], false, bhi, (short)0, acc, false, false);
        }
        // C layout: VGPR r -> row (r + 8*hh), col ln (within tile)
#pragma unroll
        for (int r = 0; r < 8; ++r) {
            int mm = row0 + 8 * hh + r;
            if (mm < M) H[(size_t)mm * DIM + nt * 16 + ln] = acc[r];
        }
    }
}

// ---------- out[m] = dot(H[m,:128], a[:128]), one wave per row ----------
__global__ void __launch_bounds__(256)
rowdot_kernel(const float* __restrict__ H, const float* __restrict__ a,
              float* __restrict__ out, int M) {
    int gw   = (int)(((long)blockIdx.x * blockDim.x + threadIdx.x) >> 5);
    int lane = threadIdx.x & 31;
    if (gw >= M) return;
    const float4* h4 = (const float4*)(H + (size_t)gw * DIM);
    const float4* a4 = (const float4*)a;
    float4 hv = h4[lane], av = a4[lane];
    float s = hv.x * av.x + hv.y * av.y + hv.z * av.z + hv.w * av.w;
#pragma unroll
    for (int off = 16; off; off >>= 1) s += __shfl_down(s, off, 32);
    if (lane == 0) out[gw] = s;
}

// ---------- edge pass 1: logits + segment max (order-preserving uint atomics) ----------
__global__ void edge_max_kernel(const int* __restrict__ oi, const int* __restrict__ ri,
                                const float* __restrict__ aS, const float* __restrict__ aD,
                                float* __restrict__ evals, unsigned* __restrict__ mord, int E) {
    int e = blockIdx.x * blockDim.x + threadIdx.x;
    if (e >= E) return;
    float v = aS[oi[e]] + aD[ri[e]];
    v = (v > 0.0f) ? v : v * 0.2f;          // leaky_relu
    evals[e] = v;
    atomicMax(&mord[ri[e]], f2ord(v));
}

// ---------- edge pass 2: exp-weighted scatter-add of hs rows; one wave per edge ----------
__global__ void __launch_bounds__(256)
edge_accum_kernel(const int* __restrict__ oi, const int* __restrict__ ri,
                  const float* __restrict__ evals, const unsigned* __restrict__ mord,
                  const float* __restrict__ hs,
                  float* __restrict__ s, float* __restrict__ acc, int E) {
    int gw   = (int)(((long)blockIdx.x * blockDim.x + threadIdx.x) >> 5);
    int lane = threadIdx.x & 31;
    if (gw >= E) return;
    const int o = oi[gw], r = ri[gw];
    const float mx = ord2f(mord[r]);
    const float ex = expf(evals[gw] - mx);
    if (lane == 0) atomicAdd(&s[r], ex);
    const float4 hv = ((const float4*)(hs + (size_t)o * DIM))[lane];
    float* dst = acc + (size_t)r * DIM + lane * 4;
    atomicAdd(dst + 0, ex * hv.x);
    atomicAdd(dst + 1, ex * hv.y);
    atomicAdd(dst + 2, ex * hv.z);
    atomicAdd(dst + 3, ex * hv.w);
}

// ---------- rider finalize: out = [relu](acc / (s+eps) + b) ----------
__global__ void rider_finalize_kernel(const float* __restrict__ acc, const float* __restrict__ s,
                                      const float* __restrict__ b, float* __restrict__ out,
                                      int n, int do_relu) {
    int i = blockIdx.x * blockDim.x + threadIdx.x;
    if (i >= n * DIM) return;
    int r = i >> 7, c = i & 127;
    float v = acc[i] / (s[r] + 1e-16f) + b[c];
    if (do_relu) v = fmaxf(v, 0.0f);
    out[i] = v;
}

// ---------- per-edge scoring: sigmoid(dot(o1[oi], r2[ri])) ----------
__global__ void __launch_bounds__(256)
score_kernel(const int* __restrict__ oi, const int* __restrict__ ri,
             const float* __restrict__ o1, const float* __restrict__ r2,
             float* __restrict__ out, int E) {
    int gw   = (int)(((long)blockIdx.x * blockDim.x + threadIdx.x) >> 5);
    int lane = threadIdx.x & 31;
    if (gw >= E) return;
    const float4 a = ((const float4*)(o1 + (size_t)oi[gw] * DIM))[lane];
    const float4 b = ((const float4*)(r2 + (size_t)ri[gw] * DIM))[lane];
    float s = a.x * b.x + a.y * b.y + a.z * b.z + a.w * b.w;
#pragma unroll
    for (int off = 16; off; off >>= 1) s += __shfl_down(s, off, 32);
    if (lane == 0) out[gw] = 1.0f / (1.0f + expf(-s));
}

// ---------- launcher ----------
static inline int cdiv(long a, long b) { return (int)((a + b - 1) / b); }

extern "C" void kernel_launch(void* const* d_in, const int* in_sizes, int n_in,
                              void* d_out, int out_size, void* d_ws, size_t ws_size,
                              hipStream_t stream) {
    const float* x_order = (const float*)d_in[0];
    const float* x_rider = (const float*)d_in[1];
    const int*   oi      = (const int*)d_in[2];
    const int*   ri      = (const int*)d_in[3];
    const float* Ws1 = (const float*)d_in[4];
    const float* Wd1 = (const float*)d_in[5];
    const float* as1 = (const float*)d_in[6];
    const float* ad1 = (const float*)d_in[7];
    const float* b1  = (const float*)d_in[8];
    const float* Ws2 = (const float*)d_in[9];
    const float* Wd2 = (const float*)d_in[10];
    const float* as2 = (const float*)d_in[11];
    const float* ad2 = (const float*)d_in[12];
    const float* b2  = (const float*)d_in[13];
    float* out = (float*)d_out;

    const int NO = in_sizes[0] / DIM;
    const int NR = in_sizes[1] / DIM;
    const int E  = in_sizes[2];

    // workspace bump allocation (all sizes multiples of 4 floats -> float4 aligned)
    float* p = (float*)d_ws;
    float*    o1    = p; p += (size_t)NO * DIM;
    float*    hs    = p; p += (size_t)NO * DIM;   // reused by both layers
    float*    alpS  = p; p += NO;
    float*    alpD  = p; p += NR;
    float*    wdv   = p; p += DIM;
    float*    evals = p; p += E;
    unsigned* mord  = (unsigned*)p; p += NR;      // contiguous with s, acc for one zero pass
    float*    svec  = p; p += NR;
    float*    acc   = p; p += (size_t)NR * DIM;
    float*    r1    = p; p += (size_t)NR * DIM;
    float*    r2    = p; p += (size_t)NR * DIM;

    const long zlen = 2L * NR + (long)NR * DIM;   // mord + svec + acc

    // o1 = relu(x_order)  (needed for layer 2 and scoring)
    relu_kernel<<<cdiv((long)NO * DIM, 256), 256, 0, stream>>>(x_order, o1, (long)NO * DIM);

    // ================= layer 1 =================
    gemm_bf16x3_kernel<<<cdiv(NO, 128), 256, 0, stream>>>(x_order, Ws1, hs, NO);
    rowdot_kernel<<<cdiv((long)NO * 32, 256), 256, 0, stream>>>(hs, as1, alpS, NO);
    rowdot_kernel<<<cdiv((long)DIM * 32, 256), 256, 0, stream>>>(Wd1, ad1, wdv, DIM);   // wdv = Wd1 @ ad1
    rowdot_kernel<<<cdiv((long)NR * 32, 256), 256, 0, stream>>>(x_rider, wdv, alpD, NR); // alpha_d = x_rider @ wdv
    zero_kernel<<<cdiv(zlen, 256), 256, 0, stream>>>((float*)mord, zlen);
    edge_max_kernel<<<cdiv(E, 256), 256, 0, stream>>>(oi, ri, alpS, alpD, evals, mord, E);
    edge_accum_kernel<<<cdiv((long)E * 32, 256), 256, 0, stream>>>(oi, ri, evals, mord, hs, svec, acc, E);
    rider_finalize_kernel<<<cdiv((long)NR * DIM, 256), 256, 0, stream>>>(acc, svec, b1, r1, NR, 1);

    // ================= layer 2 =================
    gemm_bf16x3_kernel<<<cdiv(NO, 128), 256, 0, stream>>>(o1, Ws2, hs, NO);
    rowdot_kernel<<<cdiv((long)NO * 32, 256), 256, 0, stream>>>(hs, as2, alpS, NO);
    rowdot_kernel<<<cdiv((long)DIM * 32, 256), 256, 0, stream>>>(Wd2, ad2, wdv, DIM);
    rowdot_kernel<<<cdiv((long)NR * 32, 256), 256, 0, stream>>>(r1, wdv, alpD, NR);
    zero_kernel<<<cdiv(zlen, 256), 256, 0, stream>>>((float*)mord, zlen);
    edge_max_kernel<<<cdiv(E, 256), 256, 0, stream>>>(oi, ri, alpS, alpD, evals, mord, E);
    edge_accum_kernel<<<cdiv((long)E * 32, 256), 256, 0, stream>>>(oi, ri, evals, mord, hs, svec, acc, E);
    rider_finalize_kernel<<<cdiv((long)NR * DIM, 256), 256, 0, stream>>>(acc, svec, b2, r2, NR, 0);

    // ================= scoring =================
    score_kernel<<<cdiv((long)E * 32, 256), 256, 0, stream>>>(oi, ri, o1, r2, out, E);
}
